// MultiHeadAttention_48533130445048
// MI455X (gfx1250) — matrix operand order repaired and checked by reading, not compiled, not run on hardware
//
#include <hip/hip_runtime.h>
#include <hip/hip_bf16.h>

// MI455X (gfx1250, wave32) fused multi-head attention:
//   1) Q/K projections -> f16 workspace [B*H, S, 64]
//   2) V projection    -> f16 workspace transposed [B*H, 64, S]
//   3) flash attention (causal): block-cooperative K/V^T staging to LDS via
//      GLOBAL_LOAD_ASYNC_TO_LDS_B128 (double-buffered, ASYNCcnt-paced),
//      f16 WMMA + fp32 online softmax, split masked/unmasked chunk paths
//   4) output projection -> fp32 d_out
// All matmuls via v_wmma_f32_16x16x32_f16 (f32 accumulate). Projection GEMMs
// are software-pipelined one k-step deep so VMEM clauses overlap the WMMAs.

#define DEV __device__ __forceinline__

typedef _Float16 v16h __attribute__((ext_vector_type(16)));
typedef _Float16 v8h  __attribute__((ext_vector_type(8)));
typedef float    v8f  __attribute__((ext_vector_type(8)));
typedef float    v4f  __attribute__((ext_vector_type(4)));

static constexpr int B_  = 2;
static constexpr int S_  = 2048;
static constexpr int D_  = 1024;
static constexpr int H_  = 16;
static constexpr int HD_ = 64;

DEV v8f wmma_f16(v16h a, v16h b, v8f c) {
  // (neg_a, A, neg_b, B, c_mod, C, reuse_a, reuse_b)
  return __builtin_amdgcn_wmma_f32_16x16x32_f16(false, a, false, b, (short)0, c,
                                                false, false);
}

// ---- A fragment: 16x32 (M x K). Lane m = lane&15;
//      halves 0..15 -> K = kk + 16*(lane>>4) + 0..15.
DEV v16h load_a(const float* A, int lda, int m0, int kk, int lane) {
  const float* p = A + (size_t)(m0 + (lane & 15)) * lda + kk + ((lane >> 4) << 4);
  v4f x0 = *(const v4f*)(p);
  v4f x1 = *(const v4f*)(p + 4);
  v4f x2 = *(const v4f*)(p + 8);
  v4f x3 = *(const v4f*)(p + 12);
  v16h r;
#pragma unroll
  for (int i = 0; i < 4; ++i) {
    r[i]      = (_Float16)x0[i];
    r[4 + i]  = (_Float16)x1[i];
    r[8 + i]  = (_Float16)x2[i];
    r[12 + i] = (_Float16)x3[i];
  }
  return r;
}
DEV v16h load_a(const _Float16* A, int lda, int m0, int kk, int lane) {
  const _Float16* p = A + (size_t)(m0 + (lane & 15)) * lda + kk + ((lane >> 4) << 4);
  v8h lo = *(const v8h*)(p);
  v8h hi = *(const v8h*)(p + 8);
  v16h r;
#pragma unroll
  for (int i = 0; i < 8; ++i) { r[i] = lo[i]; r[8 + i] = hi[i]; }
  return r;
}

// ---- B fragment: 32x16 (K x N) sourced from Bt (N x K row-major).
//      Lane n = lane&15; halves 0..15 -> K = kk + 16*(lane>>4) + 0..15.
DEV v16h load_bt(const float* Bt, int ldb, int n0, int kk, int lane) {
  const float* p = Bt + (size_t)(n0 + (lane & 15)) * ldb + kk + ((lane >> 4) << 4);
  v4f x0 = *(const v4f*)(p);
  v4f x1 = *(const v4f*)(p + 4);
  v4f x2 = *(const v4f*)(p + 8);
  v4f x3 = *(const v4f*)(p + 12);
  v16h r;
#pragma unroll
  for (int i = 0; i < 4; ++i) {
    r[i]      = (_Float16)x0[i];
    r[4 + i]  = (_Float16)x1[i];
    r[8 + i]  = (_Float16)x2[i];
    r[12 + i] = (_Float16)x3[i];
  }
  return r;
}
DEV v16h load_bt(const _Float16* Bt, int ldb, int n0, int kk, int lane) {
  const _Float16* p = Bt + (size_t)(n0 + (lane & 15)) * ldb + kk + ((lane >> 4) << 4);
  v8h lo = *(const v8h*)(p);
  v8h hi = *(const v8h*)(p + 8);
  v16h r;
#pragma unroll
  for (int i = 0; i < 8; ++i) { r[i] = lo[i]; r[8 + i] = hi[i]; }
  return r;
}

// ---------------- projection GEMM: out = X @ W^T + bias ----------------
// Each wave computes a 16(M) x 64(N) strip, software-pipelined one k-step
// deep: the next iteration's 20 b128 loads are issued before this iteration's
// WMMAs so the VMEM clause flies under the matrix ops. mode: 0 = f16
// head-major [bh,S,hd] (Q/K), 1 = f16 head-transposed [bh,hd,S] (V),
// 2 = fp32 row-major (final out).
template <typename AT>
__global__ __launch_bounds__(256) void proj_kernel(const AT* __restrict__ X,
                                                   const float* __restrict__ W,
                                                   const float* __restrict__ bias,
                                                   _Float16* __restrict__ outh,
                                                   float* __restrict__ outf,
                                                   int mode) {
  const int lane = threadIdx.x & 31;
  const int wid  = blockIdx.x * (blockDim.x >> 5) + (threadIdx.x >> 5);
  const int NS   = D_ / 64;  // 16 N-strips per row of tiles
  const int m0   = (wid / NS) * 16;
  const int n0   = (wid % NS) * 64;

  v8f acc[4] = {};
  // pipeline prologue: fragments for kk = 0
  v16h a  = load_a(X, D_, m0, 0, lane);
  v16h b0 = load_bt(W, D_, n0, 0, lane);
  v16h b1 = load_bt(W, D_, n0 + 16, 0, lane);
  v16h b2 = load_bt(W, D_, n0 + 32, 0, lane);
  v16h b3 = load_bt(W, D_, n0 + 48, 0, lane);
  for (int kk = 32; kk < D_; kk += 32) {
    // issue next k-step's loads before consuming the current fragments
    v16h an  = load_a(X, D_, m0, kk, lane);
    v16h bn0 = load_bt(W, D_, n0, kk, lane);
    v16h bn1 = load_bt(W, D_, n0 + 16, kk, lane);
    v16h bn2 = load_bt(W, D_, n0 + 32, kk, lane);
    v16h bn3 = load_bt(W, D_, n0 + 48, kk, lane);
    acc[0] = wmma_f16(a, b0, acc[0]);
    acc[1] = wmma_f16(a, b1, acc[1]);
    acc[2] = wmma_f16(a, b2, acc[2]);
    acc[3] = wmma_f16(a, b3, acc[3]);
    a = an; b0 = bn0; b1 = bn1; b2 = bn2; b3 = bn3;
  }
  acc[0] = wmma_f16(a, b0, acc[0]);
  acc[1] = wmma_f16(a, b1, acc[1]);
  acc[2] = wmma_f16(a, b2, acc[2]);
  acc[3] = wmma_f16(a, b3, acc[3]);

  const int colL = lane & 15;
  const int hi   = lane >> 4;
#pragma unroll
  for (int t = 0; t < 4; ++t) {
    const int e    = n0 + 16 * t + colL;       // output column
    const float bv = bias[e];
    if (mode == 2) {
#pragma unroll
      for (int r = 0; r < 8; ++r) {
        const int m = m0 + 8 * hi + r;         // output row
        outf[(size_t)m * D_ + e] = acc[t][r] + bv;
      }
    } else {
      const int h  = e >> 6;
      const int hd = e & 63;
#pragma unroll
      for (int r = 0; r < 8; ++r) {
        const int m = m0 + 8 * hi + r;
        const int b = m >> 11;                 // batch
        const int s = m & (S_ - 1);            // sequence pos
        const size_t idx =
            (mode == 0) ? ((size_t)((b * H_ + h) * S_ + s) * HD_ + hd)
                        : ((size_t)((b * H_ + h) * HD_ + hd) * S_ + s);
        outh[idx] = (_Float16)(acc[t][r] + bv);
      }
    }
  }
}

// ---------------- flash attention chunk (32 keys) ----------------
// S^T = K_chunk(32x64) @ Q^T(64x16): query = lane column, so the online
// softmax state (m, l) is one scalar per lane. O^T accumulated via
// O^T += V^T(64x32) @ P^T(32x16). MASK=true only for the causal boundary.
template <bool MASK>
DEV void attn_chunk(const _Float16* sKb, const _Float16* sVb, int c0, int qcol,
                    int hi, int lane, v16h bq0, v16h bq1, v8f (&o)[4],
                    float& m_run, float& l_run) {
  v8f s0 = {}, s1 = {};
  {
    v16h a0 = load_a(sKb, 64, 0, 0, lane);
    v16h a1 = load_a(sKb, 64, 0, 32, lane);
    v16h a2 = load_a(sKb, 64, 16, 0, lane);
    v16h a3 = load_a(sKb, 64, 16, 32, lane);
    s0 = wmma_f16(a0, bq0, s0);
    s0 = wmma_f16(a1, bq1, s0);
    s1 = wmma_f16(a2, bq0, s1);
    s1 = wmma_f16(a3, bq1, s1);
  }
  // V^T A-fragments issued early: the ds loads overlap the softmax VALU chain.
  v16h av0 = load_a(sVb, 32, 0, 0, lane);
  v16h av1 = load_a(sVb, 32, 16, 0, lane);
  v16h av2 = load_a(sVb, 32, 32, 0, lane);
  v16h av3 = load_a(sVb, 32, 48, 0, lane);

  float mloc = -1e30f;
#pragma unroll
  for (int r = 0; r < 8; ++r) {
    float x0 = s0[r] * 0.125f;  // 1/sqrt(64)
    float x1 = s1[r] * 0.125f;
    if (MASK) {
      const int k0i = c0 + 8 * hi + r;  // key of slot r, frag 0
      x0 = (k0i <= qcol) ? x0 : -1e30f;
      x1 = (k0i + 16 <= qcol) ? x1 : -1e30f;
    }
    s0[r] = x0;
    s1[r] = x1;
    mloc  = fmaxf(mloc, fmaxf(x0, x1));
  }
  mloc = fmaxf(mloc, __shfl_xor(mloc, 16));

  const float m_new = fmaxf(m_run, mloc);
  const float alpha = __expf(m_run - m_new);
  float psum = 0.f;
#pragma unroll
  for (int r = 0; r < 8; ++r) {
    const float p0 = __expf(s0[r] - m_new);
    const float p1 = __expf(s1[r] - m_new);
    s0[r] = p0;
    s1[r] = p1;
    psum += p0 + p1;
  }
  psum += __shfl_xor(psum, 16);
  l_run = l_run * alpha + psum;
  m_run = m_new;
#pragma unroll
  for (int t = 0; t < 4; ++t)
#pragma unroll
    for (int r = 0; r < 8; ++r) o[t][r] *= alpha;

  // Repack P^T (C/D layout) into a B fragment via cross-half lane swaps:
  // lane (hi,n) element j needs key 16*hi + j.
  v16h bp;
#pragma unroll
  for (int r = 0; r < 8; ++r) {
    const float x0 = __shfl_xor(s0[r], 16);
    const float x1 = __shfl_xor(s1[r], 16);
    bp[r]     = (_Float16)(hi ? x1 : s0[r]);
    bp[8 + r] = (_Float16)(hi ? s1[r] : x0);
  }
  o[0] = wmma_f16(av0, bp, o[0]);
  o[1] = wmma_f16(av1, bp, o[1]);
  o[2] = wmma_f16(av2, bp, o[2]);
  o[3] = wmma_f16(av3, bp, o[3]);
}

// ---------------- flash attention (causal, block-cooperative) ----------------
// 8 waves/block own 8 adjacent query tiles of the same head and share the K /
// V^T chunk, staged into LDS by the async copy engine, double-buffered.
__global__ __launch_bounds__(256) void attn_kernel(const _Float16* __restrict__ qws,
                                                   const _Float16* __restrict__ kws,
                                                   const _Float16* __restrict__ vtws,
                                                   _Float16* __restrict__ ows) {
  __shared__ __align__(16) _Float16 sK[2][32 * 64];  // 2 x 4 KB: 32 keys x 64 hd
  __shared__ __align__(16) _Float16 sV[2][64 * 32];  // 2 x 4 KB: 64 hd x 32 keys

  const int tid  = threadIdx.x;
  const int lane = tid & 31;
  const int wv   = tid >> 5;                 // wave in block, 0..7
  const int tile0 = blockIdx.x * 8;          // first q-tile of this block
  const int bh    = tile0 >> 7;              // 128 q-tiles per head
  const int q0    = ((tile0 & 127) + wv) * 16;
  const int qb0   = (tile0 & 127) * 16;      // block's first query
  const int nch   = (qb0 + 128) / 32;        // uniform chunk count (block max)

  const _Float16* Q  = qws + (size_t)bh * S_ * HD_;
  const _Float16* K  = kws + (size_t)bh * S_ * HD_;
  const _Float16* Vt = vtws + (size_t)bh * HD_ * S_;

  // Q^T B-fragments, loaded once from global.
  const v16h bq0 = load_bt(Q, HD_, q0, 0, lane);
  const v16h bq1 = load_bt(Q, HD_, q0, 32, lane);

  // Async-stage one 32-key chunk of K (contiguous 4 KB) and V^T (64 x 64 B
  // rows) into LDS: 256 threads x one b128 each per matrix. LDS byte offset =
  // low 32 bits of the generic pointer (ISA 10.2 aperture truncation).
  auto stage = [&](int c0, int buf) {
    const unsigned kl = (unsigned)(unsigned long long)&sK[buf][0] + (unsigned)tid * 16u;
    const unsigned long long kg =
        (unsigned long long)(K + (size_t)c0 * HD_) + (unsigned long long)tid * 16u;
    asm volatile("global_load_async_to_lds_b128 %0, %1, off" ::"v"(kl), "v"(kg)
                 : "memory");
    const int hd = tid >> 2, seg = tid & 3;
    const unsigned vl =
        (unsigned)(unsigned long long)&sV[buf][0] + (unsigned)(hd * 64 + seg * 16);
    const unsigned long long vg =
        (unsigned long long)(Vt + (size_t)hd * S_ + c0 + seg * 8);
    asm volatile("global_load_async_to_lds_b128 %0, %1, off" ::"v"(vl), "v"(vg)
                 : "memory");
  };

  v8f o[4] = {};
  float m_run = -1e30f, l_run = 0.f;
  const int qcol  = q0 + (lane & 15);
  const int hi    = lane >> 4;
  const int kendw = q0 + 16;  // this wave's causal key bound

  stage(0, 0);
  for (int ci = 0; ci < nch; ++ci) {
    const int c0 = ci * 32;
    asm volatile("s_wait_asynccnt 0" ::: "memory");  // my staged copies landed
    __syncthreads();  // chunk ci visible block-wide; chunk ci-1 fully consumed
    if (ci + 1 < nch) stage(c0 + 32, (ci + 1) & 1);  // prefetch next chunk
    if (c0 < kendw) {                                // wave-uniform causal skip
      const _Float16* sKb = &sK[ci & 1][0];
      const _Float16* sVb = &sV[ci & 1][0];
      if (c0 + 32 > q0)  // causal boundary chunk(s) only
        attn_chunk<true>(sKb, sVb, c0, qcol, hi, lane, bq0, bq1, o, m_run, l_run);
      else
        attn_chunk<false>(sKb, sVb, c0, qcol, hi, lane, bq0, bq1, o, m_run, l_run);
    }
  }

  // Normalize and write f16 [B*S, D] (row = token, col = h*64 + hd).
  const float inv  = 1.f / l_run;
  const int bidx   = bh >> 4;
  const int hidx   = bh & 15;
  const size_t row = (size_t)(bidx * S_ + qcol) * D_ + hidx * HD_;
#pragma unroll
  for (int t = 0; t < 4; ++t) {
    v8h outv;
#pragma unroll
    for (int r = 0; r < 8; ++r) outv[r] = (_Float16)(o[t][r] * inv);
    *(v8h*)(ows + row + t * 16 + 8 * hi) = outv;  // hd = t*16 + 8*hi + r
  }
}

extern "C" void kernel_launch(void* const* d_in, const int* in_sizes, int n_in,
                              void* d_out, int out_size, void* d_ws, size_t ws_size,
                              hipStream_t stream) {
  const float* query = (const float*)d_in[0];
  const float* key   = (const float*)d_in[1];
  const float* value = (const float*)d_in[2];
  // d_in[3] = boolean causal mask: handled analytically, not read.
  const float* wq = (const float*)d_in[4];
  const float* bq = (const float*)d_in[5];
  const float* wk = (const float*)d_in[6];
  const float* bk = (const float*)d_in[7];
  const float* wv = (const float*)d_in[8];
  const float* bv = (const float*)d_in[9];
  const float* wo = (const float*)d_in[10];
  const float* bo = (const float*)d_in[11];
  float* out = (float*)d_out;

  // workspace carve: 4 x 8.39 MB f16 buffers (~33.6 MB total)
  const size_t sz = (size_t)B_ * H_ * S_ * HD_;  // == B*S*D elements
  char* p = (char*)d_ws;
  _Float16* qws  = (_Float16*)p; p += sz * sizeof(_Float16);
  _Float16* kws  = (_Float16*)p; p += sz * sizeof(_Float16);
  _Float16* vtws = (_Float16*)p; p += sz * sizeof(_Float16);
  _Float16* ows  = (_Float16*)p; p += sz * sizeof(_Float16);

  const dim3 blk(256);
  // 4096 waves = (4096/16 M-tiles) x (1024/64 N-strips); 8 waves/block.
  const dim3 gproj(512);
  // 4096 q-tiles = 32 heads x 128 tiles; 8 tiles/block.
  const dim3 gattn(512);

  proj_kernel<float><<<gproj, blk, 0, stream>>>(query, wq, bq, qws, nullptr, 0);
  proj_kernel<float><<<gproj, blk, 0, stream>>>(key, wk, bk, kws, nullptr, 0);
  proj_kernel<float><<<gproj, blk, 0, stream>>>(value, wv, bv, vtws, nullptr, 1);
  attn_kernel<<<gattn, blk, 0, stream>>>(qws, kws, vtws, ows);
  proj_kernel<_Float16><<<gproj, blk, 0, stream>>>(ows, wo, bo, nullptr, out, 2);
}
